// DualExtractor_78494822302241
// MI455X (gfx1250) — compile-verified, matrix operand order checked by reading, not run on hardware
//
#include <hip/hip_runtime.h>
#include <hip/hip_bf16.h>

// ---------------------------------------------------------------------------
// Types for WMMA fragments (CDNA5 / gfx1250, wave32)
// ---------------------------------------------------------------------------
typedef __attribute__((ext_vector_type(16))) __bf16 v16bf;
typedef __attribute__((ext_vector_type(8)))  __bf16 v8bf;
typedef __attribute__((ext_vector_type(4)))  __bf16 v4bf;
typedef __attribute__((ext_vector_type(8)))  float  v8f;
typedef __attribute__((ext_vector_type(4)))  int    v4i;

#define DIM     768
#define HEADS   6
#define POINTS  4
#define HIDDEN  192
#define HKV     64
#define WKV     64
#define NQ      4096   // 64*64

// ---------------------------------------------------------------------------
// CDNA5 async global->LDS copy (GLOBAL_LOAD_ASYNC_TO_LDS_B128, ASYNCcnt).
// Falls back to a synchronous copy through the same LDS staging if the
// builtins are unavailable on this toolchain.
// ---------------------------------------------------------------------------
#if defined(__has_builtin)
#if __has_builtin(__builtin_amdgcn_global_load_async_to_lds_b128) && \
    __has_builtin(__builtin_amdgcn_s_wait_asynccnt)
#define HAVE_ASYNC_LDS 1
#endif
#endif
#ifndef HAVE_ASYNC_LDS
#define HAVE_ASYNC_LDS 0
#endif

#define AS_GLOBAL __attribute__((address_space(1)))
#define AS_SHARED __attribute__((address_space(3)))

__device__ __forceinline__
void async_copy16(const __bf16* __restrict__ g, __bf16* __restrict__ s) {
#if HAVE_ASYNC_LDS
    __builtin_amdgcn_global_load_async_to_lds_b128(
        (AS_GLOBAL v4i*)g, (AS_SHARED v4i*)s, 0, 0);
#else
    *(v8bf*)s = *(const v8bf*)g;
#endif
}

template <int N>
__device__ __forceinline__ void async_wait_le() {
#if HAVE_ASYNC_LDS
    __builtin_amdgcn_s_wait_asynccnt(N);   // immediate operand required
#endif
}

// ---------------------------------------------------------------------------
// Weight convert: f32 (K x Ncols, row-major) -> bf16 transposed (Npad x K),
// optionally packing two source matrices (for [Wo | Wa]) and zero padding.
// ---------------------------------------------------------------------------
__global__ __launch_bounds__(256)
void cvt_transpose_bf16(const float* __restrict__ src0, int N0,
                        const float* __restrict__ src1, int N1,
                        __bf16* __restrict__ dst, int K, int Npad) {
    int idx = blockIdx.x * 256 + threadIdx.x;
    if (idx >= Npad * K) return;
    int n = idx / K;
    int k = idx - n * K;
    float v = 0.0f;
    if (n < N0)            v = src0[(size_t)k * N0 + n];
    else if (n < N0 + N1)  v = src1[(size_t)k * N1 + (n - N0)];
    dst[idx] = (__bf16)v;
}

__global__ __launch_bounds__(256)
void pack_bias(const float* __restrict__ s0, int N0,
               const float* __restrict__ s1, int N1,
               float* __restrict__ dst, int Npad) {
    int i = blockIdx.x * 256 + threadIdx.x;
    if (i >= Npad) return;
    float v = 0.0f;
    if (i < N0)            v = s0[i];
    else if (i < N0 + N1)  v = s1[i - N0];
    dst[i] = v;
}

// ---------------------------------------------------------------------------
// LayerNorm over D=768, one block per row, f32 in -> bf16 out
// ---------------------------------------------------------------------------
__global__ __launch_bounds__(256)
void layernorm_bf16(const float* __restrict__ x, const float* __restrict__ g,
                    const float* __restrict__ b, __bf16* __restrict__ out) {
    const int row = blockIdx.x;
    const int t = threadIdx.x;
    const float* xr = x + (size_t)row * DIM;
    float v0 = xr[t], v1 = xr[t + 256], v2 = xr[t + 512];
    float s  = v0 + v1 + v2;
    float s2 = v0 * v0 + v1 * v1 + v2 * v2;
#pragma unroll
    for (int off = 16; off > 0; off >>= 1) {
        s  += __shfl_xor(s,  off, 32);
        s2 += __shfl_xor(s2, off, 32);
    }
    __shared__ float ss[8], ssq[8];
    const int wave = t >> 5, lane = t & 31;
    if (lane == 0) { ss[wave] = s; ssq[wave] = s2; }
    __syncthreads();
    float ts = 0.f, ts2 = 0.f;
#pragma unroll
    for (int i = 0; i < 8; ++i) { ts += ss[i]; ts2 += ssq[i]; }
    const float mean = ts * (1.0f / DIM);
    const float var  = ts2 * (1.0f / DIM) - mean * mean;
    const float r    = __frsqrt_rn(var + 1e-6f);
    __bf16* o = out + (size_t)row * DIM;
    o[t]       = (__bf16)((v0 - mean) * r * g[t]       + b[t]);
    o[t + 256] = (__bf16)((v1 - mean) * r * g[t + 256] + b[t + 256]);
    o[t + 512] = (__bf16)((v2 - mean) * r * g[t + 512] + b[t + 512]);
}

// ---------------------------------------------------------------------------
// Fragment load from an LDS tile row (32 bf16 per row = 64B).
// ISA layout (CDNA5 ISA): lanes 0-15 row M=lane, K=0..7 then 16..23;
// lanes 16-31 same rows, K=8..15 then 24..31. kh = element offset 0 or 8.
// ---------------------------------------------------------------------------
__device__ __forceinline__
v16bf lds_frag(const __bf16* rowp, int kh) {
    v8bf lo = *(const v8bf*)(rowp + kh);
    v8bf hi = *(const v8bf*)(rowp + 16 + kh);
    v16bf f;
#pragma unroll
    for (int i = 0; i < 8; ++i) { f[i] = lo[i]; f[i + 8] = hi[i]; }
    return f;
}

// ---------------------------------------------------------------------------
// WMMA GEMM:  out[M,N] = alpha*(A[M,K] @ Bt[N,K]^T + bias[N]) + resid[M,N]
//   alpha = alphaConst * (*gate if gate else 1)
//   Block = 256 thr = 8 wave32 in 2x4 -> tile 64(M) x 128(N); wave tile 32x32.
//   K mult of 32; M mult of 64; N mult of 128.
//   Tiles double-buffered in LDS via async global->LDS copies (3 per wave
//   per tile), fragments served by ds_load_b128.
// ---------------------------------------------------------------------------
__device__ __forceinline__
void store_tile(v8f acc, int m0, int n0, int lane, int N, float alpha,
                const float* __restrict__ bias, const float* __restrict__ resid,
                float* __restrict__ outF, __bf16* __restrict__ outH) {
    const int col   = n0 + (lane & 15);
    const int rbase = m0 + ((lane >> 4) << 3);
    const float bc  = bias ? bias[col] : 0.0f;
#pragma unroll
    for (int v = 0; v < 8; ++v) {
        const int row = rbase + v;
        float r = alpha * (acc[v] + bc);
        if (resid) r += resid[(size_t)row * N + col];
        if (outF) outF[(size_t)row * N + col] = r;
        else      outH[(size_t)row * N + col] = (__bf16)r;
    }
}

__global__ __launch_bounds__(256)
void gemm_bf16_wmma(const __bf16* __restrict__ A, const __bf16* __restrict__ Bt,
                    const float* __restrict__ bias, const float* __restrict__ resid,
                    const float* __restrict__ gate, float alphaConst,
                    float* __restrict__ outF, __bf16* __restrict__ outH,
                    int M, int N, int K) {
    __shared__ __bf16 smA[2][64 * 32];    // 4 KB per buffer
    __shared__ __bf16 smB[2][128 * 32];   // 8 KB per buffer
    const int t    = threadIdx.x;
    const int lane = t & 31;
    const int wave = t >> 5;
    const int wm = wave & 1;              // 0..1
    const int wn = wave >> 1;             // 0..3
    const int mBlk = blockIdx.y * 64;
    const int nBlk = blockIdx.x * 128;

    // One A chunk + two B chunks of 16B per thread per tile -> 3 async
    // instructions per wave per tile (ASYNCcnt += 3).
    const int aRow = t >> 2, aPart = t & 3;
    auto issue_tile = [&](int k0, int buf) {
        async_copy16(A + (size_t)(mBlk + aRow) * K + k0 + aPart * 8,
                     &smA[buf][aRow * 32 + aPart * 8]);
#pragma unroll
        for (int j = 0; j < 2; ++j) {
            const int ch = t + j * 256;
            const int row = ch >> 2, part = ch & 3;
            async_copy16(Bt + (size_t)(nBlk + row) * K + k0 + part * 8,
                         &smB[buf][row * 32 + part * 8]);
        }
    };

    v8f acc00 = {}, acc01 = {}, acc10 = {}, acc11 = {};
    const int T = K >> 5;
    const int kh = (lane >> 4) << 3;      // element offset 0 or 8
    const int la = lane & 15;

    issue_tile(0, 0);
    for (int i = 0; i < T; ++i) {
        const int cur = i & 1;
        if (i + 1 < T) {
            issue_tile((i + 1) << 5, cur ^ 1);
            if (i + 2 < T) {  // keep the L2 warm two tiles ahead
                __builtin_prefetch(A  + (size_t)(mBlk + aRow) * K + ((i + 2) << 5), 0, 1);
                __builtin_prefetch(Bt + (size_t)(nBlk + aRow) * K + ((i + 2) << 5), 0, 1);
            }
            async_wait_le<3>();            // tile i done; 3 in flight = tile i+1
        } else {
            async_wait_le<0>();            // final tile: drain everything
        }
        __syncthreads();                   // everyone's tile-i data visible

        v16bf a0 = lds_frag(&smA[cur][(wm * 32 +      la) * 32], kh);
        v16bf a1 = lds_frag(&smA[cur][(wm * 32 + 16 + la) * 32], kh);
        v16bf b0 = lds_frag(&smB[cur][(wn * 32 +      la) * 32], kh);
        v16bf b1 = lds_frag(&smB[cur][(wn * 32 + 16 + la) * 32], kh);
        acc00 = __builtin_amdgcn_wmma_f32_16x16x32_bf16(false, a0, false, b0, (short)0, acc00, false, false);
        acc01 = __builtin_amdgcn_wmma_f32_16x16x32_bf16(false, a0, false, b1, (short)0, acc01, false, false);
        acc10 = __builtin_amdgcn_wmma_f32_16x16x32_bf16(false, a1, false, b0, (short)0, acc10, false, false);
        acc11 = __builtin_amdgcn_wmma_f32_16x16x32_bf16(false, a1, false, b1, (short)0, acc11, false, false);

        __syncthreads();                   // buffer free for overwrite
    }

    const float alpha = alphaConst * (gate ? gate[0] : 1.0f);
    const int m0 = mBlk + wm * 32;
    const int n0 = nBlk + wn * 32;
    store_tile(acc00, m0,      n0,      lane, N, alpha, bias, resid, outF, outH);
    store_tile(acc01, m0,      n0 + 16, lane, N, alpha, bias, resid, outF, outH);
    store_tile(acc10, m0 + 16, n0,      lane, N, alpha, bias, resid, outF, outH);
    store_tile(acc11, m0 + 16, n0 + 16, lane, N, alpha, bias, resid, outF, outH);
}

// ---------------------------------------------------------------------------
// Deformable sampling: one wave per (row, head).
//   offaw: (M,128) f32  [0:48) offsets head-major (h*8 + p*2 + {x,y}),
//                       [48:72) attn logits (h*4 + p)
//   v:     (B*4096, 768) bf16 (head slice h*128..)   -- L2-resident (25 MB)
//   out:   (M, 768) bf16 sampled features
// ---------------------------------------------------------------------------
__global__ __launch_bounds__(256)
void msd_sample(const float* __restrict__ offaw, const __bf16* __restrict__ v,
                __bf16* __restrict__ out) {
    const int lane = threadIdx.x & 31;
    const int gw   = (blockIdx.x << 3) + (threadIdx.x >> 5);
    const int head = gw % HEADS;
    const int row  = gw / HEADS;              // b*4096 + q
    const int b    = row >> 12;
    const int q    = row & (NQ - 1);
    const float qx = (float)(q & 63);
    const float qy = (float)(q >> 6);

    const float* oa = offaw + (size_t)row * 128;
    float l0 = oa[48 + head * 4 + 0];
    float l1 = oa[48 + head * 4 + 1];
    float l2 = oa[48 + head * 4 + 2];
    float l3 = oa[48 + head * 4 + 3];
    float mx = fmaxf(fmaxf(l0, l1), fmaxf(l2, l3));
    float e0 = __expf(l0 - mx), e1 = __expf(l1 - mx);
    float e2 = __expf(l2 - mx), e3 = __expf(l3 - mx);
    float inv = 1.0f / (e0 + e1 + e2 + e3);
    float aw[POINTS] = { e0 * inv, e1 * inv, e2 * inv, e3 * inv };

    float a0 = 0.f, a1 = 0.f, a2 = 0.f, a3 = 0.f;
    const __bf16* vb = v + (size_t)b * NQ * DIM + head * 128 + lane * 4;

#pragma unroll
    for (int p = 0; p < POINTS; ++p) {
        const float ox = oa[head * 8 + p * 2 + 0];
        const float oy = oa[head * 8 + p * 2 + 1];
        // px = ((qx+0.5)/Wq + ox/Wkv)*Wkv - 0.5 with Wq==Wkv==64  ->  qx + ox
        const float px = qx + ox;
        const float py = qy + oy;
        const float x0f = floorf(px), y0f = floorf(py);
        const float wx = px - x0f, wy = py - y0f;
        const int x0 = (int)x0f, y0 = (int)y0f;
        const float cw[4] = { (1.f - wx) * (1.f - wy) * aw[p],
                              wx * (1.f - wy) * aw[p],
                              (1.f - wx) * wy * aw[p],
                              wx * wy * aw[p] };
        const int cx[4] = { x0, x0 + 1, x0,     x0 + 1 };
        const int cy[4] = { y0, y0,     y0 + 1, y0 + 1 };
#pragma unroll
        for (int cI = 0; cI < 4; ++cI) {
            const int xi = cx[cI], yi = cy[cI];
            if (xi >= 0 && xi < WKV && yi >= 0 && yi < HKV) {
                const float w = cw[cI];
                v4bf d = *(const v4bf*)(vb + (size_t)(yi * WKV + xi) * DIM);
                a0 += w * (float)d[0];
                a1 += w * (float)d[1];
                a2 += w * (float)d[2];
                a3 += w * (float)d[3];
            }
        }
    }
    __bf16* o = out + (size_t)row * DIM + head * 128 + lane * 4;
    o[0] = (__bf16)a0; o[1] = (__bf16)a1; o[2] = (__bf16)a2; o[3] = (__bf16)a3;
}

// ---------------------------------------------------------------------------
// Depthwise 3x3 conv (SAME, zero pad) + exact GELU; h1 f32 (M,256-padded,
// first 192 valid) -> h2 bf16 (M,192). One block per spatial pos, 192 thr.
// ---------------------------------------------------------------------------
__global__ __launch_bounds__(192)
void dwconv_gelu(const float* __restrict__ h1, const float* __restrict__ w,
                 const float* __restrict__ bias, __bf16* __restrict__ out) {
    const int c  = threadIdx.x;           // 0..191
    const int sp = blockIdx.x;            // b*4096 + y*64 + x
    const int bI = sp >> 12;
    const int y  = (sp >> 6) & 63;
    const int x  = sp & 63;
    float s = bias[c];
#pragma unroll
    for (int dy = -1; dy <= 1; ++dy) {
#pragma unroll
        for (int dx = -1; dx <= 1; ++dx) {
            const int yy = y + dy, xx = x + dx;
            if (yy >= 0 && yy < 64 && xx >= 0 && xx < 64) {
                s += h1[((size_t)(bI * NQ + yy * 64 + xx)) * 256 + c] *
                     w[c * 9 + (dy + 1) * 3 + (dx + 1)];
            }
        }
    }
    const float g = 0.5f * s * (1.0f + erff(s * 0.70710678118654752f));
    out[(size_t)sp * HIDDEN + c] = (__bf16)g;
}

// ---------------------------------------------------------------------------
// Host-side orchestration
// ---------------------------------------------------------------------------
extern "C" void kernel_launch(void* const* d_in, const int* in_sizes, int n_in,
                              void* d_out, int out_size, void* d_ws, size_t ws_size,
                              hipStream_t stream) {
    const int M = in_sizes[0] / DIM;           // B*4096 = 16384

    const float* c      = (const float*)d_in[0];
    const float* x_vit  = (const float*)d_in[1];
    const float* y_vit  = (const float*)d_in[2];
    const float* qn_g   = (const float*)d_in[3];
    const float* qn_b   = (const float*)d_in[4];
    const float* fnx_g  = (const float*)d_in[5];
    const float* fnx_b  = (const float*)d_in[6];
    const float* fny_g  = (const float*)d_in[7];
    const float* fny_b  = (const float*)d_in[8];
    const float* fn_g   = (const float*)d_in[9];
    const float* fn_b   = (const float*)d_in[10];
    const float* xo_w = (const float*)d_in[11]; const float* xo_b = (const float*)d_in[12];
    const float* xa_w = (const float*)d_in[13]; const float* xa_b = (const float*)d_in[14];
    const float* xv_w = (const float*)d_in[15]; const float* xv_b = (const float*)d_in[16];
    const float* xp_w = (const float*)d_in[17]; const float* xp_b = (const float*)d_in[18];
    const float* yo_w = (const float*)d_in[19]; const float* yo_b = (const float*)d_in[20];
    const float* ya_w = (const float*)d_in[21]; const float* ya_b = (const float*)d_in[22];
    const float* yv_w = (const float*)d_in[23]; const float* yv_b = (const float*)d_in[24];
    const float* yp_w = (const float*)d_in[25]; const float* yp_b = (const float*)d_in[26];
    const float* gate_x = (const float*)d_in[27];
    const float* gate_y = (const float*)d_in[28];
    const float* fc1_w  = (const float*)d_in[29]; const float* fc1_b = (const float*)d_in[30];
    const float* dw_w   = (const float*)d_in[31]; const float* dw_b  = (const float*)d_in[32];
    const float* fc2_w  = (const float*)d_in[33]; const float* fc2_b = (const float*)d_in[34];

    // ---- workspace layout (bytes, 256-aligned) ----
    char* ws = (char*)d_ws;
    size_t off = 0;
    auto alloc = [&](size_t bytes) -> char* {
        char* p = ws + off;
        off += (bytes + 255) & ~(size_t)255;
        return p;
    };
    const size_t actBF  = (size_t)M * DIM * sizeof(__bf16);        // 25.2 MB
    __bf16* q_bf   = (__bf16*)alloc(actBF);     // reused later as LN(c_mid)
    __bf16* xln    = (__bf16*)alloc(actBF);     // reused as sampled_x
    __bf16* yln    = (__bf16*)alloc(actBF);     // reused as sampled_y
    __bf16* v_x    = (__bf16*)alloc(actBF);
    __bf16* v_y    = (__bf16*)alloc(actBF);
    float*  offaw_x = (float*)alloc((size_t)M * 128 * sizeof(float));
    float*  offaw_y = (float*)alloc((size_t)M * 128 * sizeof(float));
    float*  h1      = (float*)offaw_x;          // reuse (M*256 f32 == 2 * M*128)
    float*  c_mid   = (float*)alloc((size_t)M * DIM * sizeof(float));
    __bf16* h2      = (__bf16*)alloc((size_t)M * HIDDEN * sizeof(__bf16));
    __bf16* wvxT = (__bf16*)alloc((size_t)DIM * DIM * sizeof(__bf16));
    __bf16* wvyT = (__bf16*)alloc((size_t)DIM * DIM * sizeof(__bf16));
    __bf16* wpxT = (__bf16*)alloc((size_t)DIM * DIM * sizeof(__bf16));
    __bf16* wpyT = (__bf16*)alloc((size_t)DIM * DIM * sizeof(__bf16));
    __bf16* owxT = (__bf16*)alloc((size_t)128 * DIM * sizeof(__bf16));
    __bf16* owyT = (__bf16*)alloc((size_t)128 * DIM * sizeof(__bf16));
    __bf16* fc1T = (__bf16*)alloc((size_t)256 * DIM * sizeof(__bf16));
    __bf16* fc2T = (__bf16*)alloc((size_t)DIM * HIDDEN * sizeof(__bf16));
    float*  obx  = (float*)alloc(128 * sizeof(float));
    float*  oby  = (float*)alloc(128 * sizeof(float));
    float*  f1b  = (float*)alloc(256 * sizeof(float));
    __bf16* cln  = q_bf;
    __bf16* samp_x = xln;
    __bf16* samp_y = yln;
    float*  outF = (float*)d_out;

    // ---- 1. weight conversion (f32 -> transposed/padded bf16) ----
    auto cvt = [&](const float* s0, int n0, const float* s1, int n1,
                   __bf16* dst, int K, int Npad) {
        int tot = Npad * K;
        cvt_transpose_bf16<<<(tot + 255) / 256, 256, 0, stream>>>(s0, n0, s1, n1, dst, K, Npad);
    };
    cvt(xv_w, DIM, nullptr, 0, wvxT, DIM, DIM);
    cvt(yv_w, DIM, nullptr, 0, wvyT, DIM, DIM);
    cvt(xp_w, DIM, nullptr, 0, wpxT, DIM, DIM);
    cvt(yp_w, DIM, nullptr, 0, wpyT, DIM, DIM);
    cvt(xo_w, HEADS * POINTS * 2, xa_w, HEADS * POINTS, owxT, DIM, 128);
    cvt(yo_w, HEADS * POINTS * 2, ya_w, HEADS * POINTS, owyT, DIM, 128);
    cvt(fc1_w, HIDDEN, nullptr, 0, fc1T, DIM, 256);
    cvt(fc2_w, DIM, nullptr, 0, fc2T, HIDDEN, DIM);
    pack_bias<<<1, 256, 0, stream>>>(xo_b, HEADS * POINTS * 2, xa_b, HEADS * POINTS, obx, 128);
    pack_bias<<<1, 256, 0, stream>>>(yo_b, HEADS * POINTS * 2, ya_b, HEADS * POINTS, oby, 128);
    pack_bias<<<1, 256, 0, stream>>>(fc1_b, HIDDEN, nullptr, 0, f1b, 256);

    // ---- 2. layer norms -> bf16 ----
    layernorm_bf16<<<M, 256, 0, stream>>>(c,     qn_g,  qn_b,  q_bf);
    layernorm_bf16<<<M, 256, 0, stream>>>(x_vit, fnx_g, fnx_b, xln);
    layernorm_bf16<<<M, 256, 0, stream>>>(y_vit, fny_g, fny_b, yln);

    auto gemm = [&](const __bf16* A, const __bf16* Bt, const float* bias,
                    const float* resid, const float* gate, float alpha,
                    float* oF, __bf16* oH, int N, int K) {
        dim3 grid(N / 128, M / 64);
        gemm_bf16_wmma<<<grid, 256, 0, stream>>>(A, Bt, bias, resid, gate, alpha,
                                                 oF, oH, M, N, K);
    };

    // ---- 3. value projections ----
    gemm(xln, wvxT, xv_b, nullptr, nullptr, 1.0f, nullptr, v_x, DIM, DIM);
    gemm(yln, wvyT, yv_b, nullptr, nullptr, 1.0f, nullptr, v_y, DIM, DIM);

    // ---- 4. offset + attention-logit projections ----
    gemm(q_bf, owxT, obx, nullptr, nullptr, 1.0f, offaw_x, nullptr, 128, DIM);
    gemm(q_bf, owyT, oby, nullptr, nullptr, 1.0f, offaw_y, nullptr, 128, DIM);

    // ---- 5. deformable sampling (one wave per row*head) ----
    {
        int waves = M * HEADS;
        msd_sample<<<waves / 8, 256, 0, stream>>>(offaw_x, v_x, samp_x);
        msd_sample<<<waves / 8, 256, 0, stream>>>(offaw_y, v_y, samp_y);
    }

    // ---- 6. output projections, fused gated residual: c_mid = c + gx*ux + gy*uy
    gemm(samp_x, wpxT, xp_b, c,     gate_x, 1.0f, c_mid, nullptr, DIM, DIM);
    gemm(samp_y, wpyT, yp_b, c_mid, gate_y, 1.0f, c_mid, nullptr, DIM, DIM);

    // ---- 7. FFN: LN -> fc1 -> dwconv+gelu -> fc2 (+residual into d_out) ----
    layernorm_bf16<<<M, 256, 0, stream>>>(c_mid, fn_g, fn_b, cln);
    gemm(cln, fc1T, f1b, nullptr, nullptr, 1.0f, h1, nullptr, 256, DIM);
    dwconv_gelu<<<M, 192, 0, stream>>>(h1, dw_w, dw_b, h2);
    gemm(h2, fc2T, fc2_b, c_mid, nullptr, 1.0f, outF, nullptr, DIM, HIDDEN);
}
